// GATK_layer_18013092839767
// MI455X (gfx1250) — matrix operand order repaired and checked by reading, not compiled, not run on hardware
//
#include <hip/hip_runtime.h>
#include <hip/hip_bf16.h>

// ---------------- problem constants (match reference) ----------------
#define NN     50000
#define EE     800000
#define HOPS_  3
#define HEADS_ 8
#define HID_   256
#define OUTC_  32
#define ET_    (EE + NN)          // edges + self loops
#define NEG_SLOPE_ 0.2f

typedef _Float16 v16h __attribute__((ext_vector_type(16)));
typedef _Float16 v8h  __attribute__((ext_vector_type(8)));
typedef float    v8f  __attribute__((ext_vector_type(8)));

union V16u { v16h v; v8h h[2]; };

// order-preserving float <-> uint encoding for atomic max
__device__ __forceinline__ unsigned enc_f32(float f) {
    unsigned u = __float_as_uint(f);
    return (u & 0x80000000u) ? ~u : (u | 0x80000000u);
}
__device__ __forceinline__ float dec_f32(unsigned u) {
    return (u & 0x80000000u) ? __uint_as_float(u & 0x7FFFFFFFu)
                             : __uint_as_float(~u);
}
#define ENC_NEG_FLT_MAX 0x00800000u   // enc(-FLT_MAX)

__device__ __forceinline__ float lrelu(float v, float s) { return v > 0.f ? v : s * v; }

// ---------------- WMMA core: one wave computes a 16x64 strip (4 N-tiles),
// reusing the A fragment across 4 accumulators.
// A: row-major f16 [rows x 256]; Bt: f16 [256 cols x 256 K] (K contiguous per col)
__device__ __forceinline__ void wmma_strip_16x64(const _Float16* __restrict__ A,
                                                 const _Float16* __restrict__ Bt,
                                                 int row0, int col0, v8f c[4]) {
    const int lane = threadIdx.x & 31;
    const int l16  = lane & 15;
    const int hsel = lane >> 4;
    const int koff = hsel * 16;   // A K-offset within 32-block
    const int nb   = hsel * 16;   // B K-offset within 32-block
    const _Float16* ap = A  + (size_t)(row0 + l16) * HID_;
    const _Float16* bp = Bt + (size_t)(col0 + l16) * HID_ + nb;
    c[0] = v8f{}; c[1] = v8f{}; c[2] = v8f{}; c[3] = v8f{};
#pragma unroll
    for (int kb = 0; kb < 8; ++kb) {
        const int kbase = kb * 32;
        V16u a;
        a.h[0] = *(const v8h*)(ap + kbase + koff);        // K = koff..koff+7
        a.h[1] = *(const v8h*)(ap + kbase + koff + 8);    // K = koff+8..koff+15
#pragma unroll
        for (int t = 0; t < 4; ++t) {
            v16h b = *(const v16h*)(bp + (size_t)t * 16 * HID_ + kbase);
            c[t] = __builtin_amdgcn_wmma_f32_16x16x32_f16(false, a.v, false, b,
                                                          (short)0, c[t], false, false);
        }
    }
}

// ---------------- kernels ----------------

// float4 -> 4x f16 convert (x is read 3 times; convert once)
__global__ void k_cvt_f16_4(const float* __restrict__ in, _Float16* __restrict__ out, int n4) {
    int t = blockIdx.x * blockDim.x + threadIdx.x;
    if (t >= n4) return;
    float4 v = ((const float4*)in)[t];
    _Float16* o = out + (size_t)t * 4;
    o[0] = (_Float16)v.x; o[1] = (_Float16)v.y; o[2] = (_Float16)v.z; o[3] = (_Float16)v.w;
}

// transpose-convert 256x256 f32 (row-major [K][M]) -> f16 [M][K]
__global__ void k_tr_w(const float* __restrict__ W, _Float16* __restrict__ Wt) {
    int t = blockIdx.x * blockDim.x + threadIdx.x;   // 65536 threads
    int k = t >> 8, m = t & 255;
    Wt[(size_t)m * 256 + k] = (_Float16)W[(size_t)k * 256 + m];
}

__global__ void k_fill_f32(float* __restrict__ p, float v, int n) {
    int t = blockIdx.x * blockDim.x + threadIdx.x;
    if (t < n) p[t] = v;
}
__global__ void k_fill_u32(unsigned* __restrict__ p, unsigned v, int n) {
    int t = blockIdx.x * blockDim.x + threadIdx.x;
    if (t < n) p[t] = v;
}

// GEMM: h = x_f16 @ Wt^T   (plain f32 store); 12500 waves, 16x64 strip each
__global__ void k_gemm(const _Float16* __restrict__ A, const _Float16* __restrict__ Bt,
                       float* __restrict__ Out) {
    int wave = blockIdx.x * (blockDim.x >> 5) + (threadIdx.x >> 5);
    if (wave >= (NN / 16) * 4) return;               // wave-uniform guard
    int tm = wave >> 2, tg = wave & 3;
    int row0 = tm * 16, col0 = tg * 64;
    v8f c[4];
    wmma_strip_16x64(A, Bt, row0, col0, c);
    int lane = threadIdx.x & 31, l16 = lane & 15, hsel = lane >> 4;
#pragma unroll
    for (int t = 0; t < 4; ++t) {
        int col = col0 + t * 16 + l16;
#pragma unroll
        for (int r = 0; r < 8; ++r)
            Out[(size_t)(row0 + r + 8 * hsel) * HID_ + col] = c[t][r];
    }
}

// Decoder GEMM + epilogue: out (+)= decay[hop]*lrelu(A@Bt^T + bias, 0.01)
__global__ void k_gemm_dec(const _Float16* __restrict__ A, const _Float16* __restrict__ Bt,
                           const float* __restrict__ bias, const float* __restrict__ decay,
                           int hop, float* __restrict__ Out) {
    int wave = blockIdx.x * (blockDim.x >> 5) + (threadIdx.x >> 5);
    if (wave >= (NN / 16) * 4) return;               // wave-uniform guard
    int tm = wave >> 2, tg = wave & 3;
    int row0 = tm * 16, col0 = tg * 64;
    v8f c[4];
    wmma_strip_16x64(A, Bt, row0, col0, c);
    int lane = threadIdx.x & 31, l16 = lane & 15, hsel = lane >> 4;
    float dk = decay[hop];
#pragma unroll
    for (int t = 0; t < 4; ++t) {
        int col = col0 + t * 16 + l16;
        float bv = bias[col];
#pragma unroll
        for (int r = 0; r < 8; ++r) {
            float v = lrelu(c[t][r] + bv, 0.01f) * dk;
            size_t o = (size_t)(row0 + r + 8 * hsel) * HID_ + col;
            if (hop == 0) Out[o] = v; else Out[o] += v;
        }
    }
}

// per-(node,head) attention logits: al = sum_c h[n,hd,c]*att[hd,c]
__global__ void k_al(const float* __restrict__ h, const float* __restrict__ asrc,
                     const float* __restrict__ adst, float* __restrict__ al_s,
                     float* __restrict__ al_d) {
    int t = blockIdx.x * blockDim.x + threadIdx.x;
    if (t >= NN * HEADS_) return;
    int n = t >> 3, hd = t & 7;
    const float* hp = h + (size_t)n * HID_ + hd * OUTC_;
    const float* as = asrc + hd * OUTC_;
    const float* ad = adst + hd * OUTC_;
    float s = 0.f, d = 0.f;
#pragma unroll 8
    for (int c = 0; c < OUTC_; ++c) { float v = hp[c]; s += v * as[c]; d += v * ad[c]; }
    al_s[t] = s; al_d[t] = d;
}

// pass 1: segment max over dst (ordered-uint atomicMax)
__global__ void k_edge_max(const int* __restrict__ src, const int* __restrict__ dst,
                           const float* __restrict__ al_s, const float* __restrict__ al_d,
                           unsigned* __restrict__ emax) {
    int t = blockIdx.x * blockDim.x + threadIdx.x;
    if (t >= ET_ * HEADS_) return;
    int e = t >> 3, hd = t & 7;
    int s, d;
    if (e < EE) { s = src[e]; d = dst[e]; } else { s = d = e - EE; }
    float v = lrelu(al_s[s * 8 + hd] + al_d[d * 8 + hd], NEG_SLOPE_);
    atomicMax(&emax[d * 8 + hd], enc_f32(v));
}

// pass 2: denom = segment_sum(exp(e - emax[dst]))
__global__ void k_edge_sum(const int* __restrict__ src, const int* __restrict__ dst,
                           const float* __restrict__ al_s, const float* __restrict__ al_d,
                           const unsigned* __restrict__ emax, float* __restrict__ den) {
    int t = blockIdx.x * blockDim.x + threadIdx.x;
    if (t >= ET_ * HEADS_) return;
    int e = t >> 3, hd = t & 7;
    int s, d;
    if (e < EE) { s = src[e]; d = dst[e]; } else { s = d = e - EE; }
    float v = lrelu(al_s[s * 8 + hd] + al_d[d * 8 + hd], NEG_SLOPE_);
    float ex = __expf(v - dec_f32(emax[d * 8 + hd]));
    atomicAdd(&den[d * 8 + hd], ex);
}

// pass 3: one wave per edge, scatter alpha-weighted messages (256 channels)
__global__ void k_edge_msg(const int* __restrict__ src, const int* __restrict__ dst,
                           const float* __restrict__ al_s, const float* __restrict__ al_d,
                           const unsigned* __restrict__ emax, const float* __restrict__ den,
                           const float* __restrict__ h, float* __restrict__ out) {
    int wave = blockIdx.x * (blockDim.x >> 5) + (threadIdx.x >> 5);
    if (wave >= ET_) return;
    int lane = threadIdx.x & 31;
    int s, d;
    if (wave < EE) { s = src[wave]; d = dst[wave]; } else { s = d = wave - EE; }
    const float* hs = h + (size_t)s * HID_;
    float* od = out + (size_t)d * HID_;
#pragma unroll
    for (int i = 0; i < HEADS_; ++i) {
        // uniform per-wave loads (same address in every lane -> broadcast)
        float v  = lrelu(al_s[s * 8 + i] + al_d[d * 8 + i], NEG_SLOPE_);
        float ex = __expf(v - dec_f32(emax[d * 8 + i]));
        float alpha = ex / (den[d * 8 + i] + 1e-16f);
        int c = i * 32 + lane;
        atomicAdd(&od[c], hs[c] * alpha);
    }
}

// bias add + f16 convert (decoder A-operand), float4-vectorized
__global__ void k_bias_cvt4(const float* __restrict__ gout, const float* __restrict__ bias,
                            _Float16* __restrict__ a_h) {
    int t = blockIdx.x * blockDim.x + threadIdx.x;   // NN*HID/4 threads
    float4 v = ((const float4*)gout)[t];
    int m = (t * 4) & 255;
    _Float16* o = a_h + (size_t)t * 4;
    o[0] = (_Float16)(v.x + bias[m + 0]);
    o[1] = (_Float16)(v.y + bias[m + 1]);
    o[2] = (_Float16)(v.z + bias[m + 2]);
    o[3] = (_Float16)(v.w + bias[m + 3]);
}

// ---------------- host launcher ----------------
extern "C" void kernel_launch(void* const* d_in, const int* in_sizes, int n_in,
                              void* d_out, int out_size, void* d_ws, size_t ws_size,
                              hipStream_t stream) {
    (void)in_sizes; (void)n_in; (void)out_size; (void)ws_size;
    const float* x     = (const float*)d_in[0];
    const int*   ei    = (const int*)  d_in[1];
    const float* gat_W = (const float*)d_in[2];
    const float* att_s = (const float*)d_in[3];
    const float* att_d = (const float*)d_in[4];
    const float* gat_b = (const float*)d_in[5];
    const float* dec_W = (const float*)d_in[6];
    const float* dec_b = (const float*)d_in[7];
    const float* decay = (const float*)d_in[8];
    float* out = (float*)d_out;

    // carve workspace (all L2-resident on MI455X: ~160 MB < 192 MB L2)
    char* ws = (char*)d_ws;
    auto carve = [&](size_t bytes) { char* p = ws; ws += (bytes + 255) & ~(size_t)255; return p; };
    _Float16* x_h  = (_Float16*)carve((size_t)NN * HID_ * 2);
    float*    hbuf = (float*)   carve((size_t)NN * HID_ * 4);
    float*    gout = (float*)   carve((size_t)NN * HID_ * 4);
    _Float16* a_h  = (_Float16*)carve((size_t)NN * HID_ * 2);
    _Float16* Wt   = (_Float16*)carve((size_t)HID_ * HID_ * 2);
    _Float16* dWt  = (_Float16*)carve((size_t)HID_ * HID_ * 2);
    float*    al_s = (float*)   carve((size_t)NN * HEADS_ * 4);
    float*    al_d = (float*)   carve((size_t)NN * HEADS_ * 4);
    unsigned* emax = (unsigned*)carve((size_t)NN * HEADS_ * 4);
    float*    den  = (float*)   carve((size_t)NN * HEADS_ * 4);

    const int B = 256;
    const int n_elem  = NN * HID_;                    // 12,800,000
    const int nh      = NN * HEADS_;                  // 400,000
    const int gemm_wv = (NN / 16) * 4;                // 12,500 strip-waves
    const int gemm_bl = (gemm_wv + 7) / 8;            // 8 waves per 256-thread block
    const int msg_bl  = (ET_ + 7) / 8;                // 1 wave per edge

    k_cvt_f16_4<<<(n_elem / 4 + B - 1) / B, B, 0, stream>>>(x, x_h, n_elem / 4);

    for (int k = 0; k < HOPS_; ++k) {
        const int* src = ei + (size_t)k * 2 * EE;
        const int* dst = src + EE;

        k_tr_w<<<HID_ * HID_ / B, B, 0, stream>>>(gat_W + (size_t)k * HID_ * HID_, Wt);
        k_tr_w<<<HID_ * HID_ / B, B, 0, stream>>>(dec_W + (size_t)k * HID_ * HID_, dWt);

        k_gemm<<<gemm_bl, B, 0, stream>>>(x_h, Wt, hbuf);

        k_al<<<(nh + B - 1) / B, B, 0, stream>>>(hbuf, att_s + k * HEADS_ * OUTC_,
                                                 att_d + k * HEADS_ * OUTC_, al_s, al_d);

        k_fill_u32<<<(nh + B - 1) / B, B, 0, stream>>>(emax, ENC_NEG_FLT_MAX, nh);
        k_fill_f32<<<(nh + B - 1) / B, B, 0, stream>>>(den, 0.f, nh);
        k_fill_f32<<<(n_elem + B - 1) / B, B, 0, stream>>>(gout, 0.f, n_elem);

        const int et_thr = ET_ * HEADS_;
        k_edge_max<<<(et_thr + B - 1) / B, B, 0, stream>>>(src, dst, al_s, al_d, emax);
        k_edge_sum<<<(et_thr + B - 1) / B, B, 0, stream>>>(src, dst, al_s, al_d, emax, den);
        k_edge_msg<<<msg_bl, B, 0, stream>>>(src, dst, al_s, al_d, emax, den, hbuf, gout);

        k_bias_cvt4<<<(n_elem / 4 + B - 1) / B, B, 0, stream>>>(gout, gat_b + k * HID_, a_h);

        k_gemm_dec<<<gemm_bl, B, 0, stream>>>(a_h, dWt, dec_b + k * HID_, decay, k, out);
    }
}
